// CrossNet_66391604462465
// MI455X (gfx1250) — matrix operand order repaired and checked by reading, not compiled, not run on hardware
//
#include <hip/hip_runtime.h>
#include <hip/hip_bf16.h>

// ---------------------------------------------------------------------------
// CrossNet axial attention for gfx1250 (MI455X, wave32, WMMA).
// All GEMM-shaped math uses v_wmma_f32_16x16x32_f16 with f32 accumulators.
// ---------------------------------------------------------------------------

#define Bn   32
#define Ln   64
#define Cn   128
#define Hn   4
#define HDn  32
#define NWIN (Bn * Ln)          // 2048 windows per axis
#define SCALE_ATTN 0.17677669529663687f   // 1/sqrt(32)

typedef __attribute__((ext_vector_type(16))) _Float16 v16h;
typedef __attribute__((ext_vector_type(8)))  float    v8f;

// ---- WMMA fragment helpers (contiguous k = 16*(lane/16) + e, wave32) ----------

// A fragment, 16x32 f16: lane holds row (base_m + lane&15); 16 contiguous
// halves at K offset ko = base_k + (lane>=16 ? 16 : 0).
__device__ __forceinline__ v16h wmma_load_a(const _Float16* s, int lda,
                                            int base_m, int base_k, int lane) {
  int m = base_m + (lane & 15);
  const _Float16* p = s + m * lda + base_k + ((lane >> 4) << 4);
  v16h a;
#pragma unroll
  for (int e = 0; e < 8; ++e) a[e] = p[e];
#pragma unroll
  for (int e = 0; e < 8; ++e) a[8 + e] = p[8 + e];
  return a;
}

// B fragment where the source is laid out [n][k] with k contiguous
// (e.g. weight row, or K-matrix row when computing Q*K^T).
// caller passes p = base + n_lane*ld + k0_lane  (k0_lane = (lane>=16?16:0)).
__device__ __forceinline__ v16h load16h(const _Float16* p) {
  v16h r;
#pragma unroll
  for (int e = 0; e < 16; ++e) r[e] = p[e];
  return r;
}

// B fragment where source is [k][n] with n contiguous (e.g. V for P@V):
// lane fixes n, walks k with stride ld.
__device__ __forceinline__ v16h load16h_strided(const _Float16* p, int ld) {
  v16h r;
#pragma unroll
  for (int e = 0; e < 16; ++e) r[e] = p[e * ld];
  return r;
}

__device__ __forceinline__ v8f wmma_f16(v16h a, v16h b, v8f c) {
  return __builtin_amdgcn_wmma_f32_16x16x32_f16(false, a, false, b,
                                                (short)0, c, false, false);
}

__device__ __forceinline__ v8f zero8() {
  v8f z = {0.f, 0.f, 0.f, 0.f, 0.f, 0.f, 0.f, 0.f};
  return z;
}

// Softmax of a 16x64 score block held as 4 D-fragments (f32), result
// (probabilities, f16) written to P[16][ldp]. Row r of the block lives in
// element r of each v8f for rows (lane>=16?8:0)+r; columns are lane&15 per
// 16-wide tile, so row reductions are shfl-xor over the 16-lane half-wave.
__device__ __forceinline__ void softmax16x64(v8f* sacc, _Float16* P, int ldp,
                                             int lane) {
  int hh = lane >> 4, ln = lane & 15;
#pragma unroll
  for (int r = 0; r < 8; ++r) {
    float s0 = sacc[0][r] * SCALE_ATTN;
    float s1 = sacc[1][r] * SCALE_ATTN;
    float s2 = sacc[2][r] * SCALE_ATTN;
    float s3 = sacc[3][r] * SCALE_ATTN;
    float m = fmaxf(fmaxf(s0, s1), fmaxf(s2, s3));
    m = fmaxf(m, __shfl_xor(m, 1));
    m = fmaxf(m, __shfl_xor(m, 2));
    m = fmaxf(m, __shfl_xor(m, 4));
    m = fmaxf(m, __shfl_xor(m, 8));
    float e0 = __expf(s0 - m), e1 = __expf(s1 - m);
    float e2 = __expf(s2 - m), e3 = __expf(s3 - m);
    float sum = e0 + e1 + e2 + e3;
    sum += __shfl_xor(sum, 1);
    sum += __shfl_xor(sum, 2);
    sum += __shfl_xor(sum, 4);
    sum += __shfl_xor(sum, 8);
    float inv = 1.0f / sum;
    int row = hh * 8 + r;
    P[row * ldp + 0 * 16 + ln] = (_Float16)(e0 * inv);
    P[row * ldp + 1 * 16 + ln] = (_Float16)(e1 * inv);
    P[row * ldp + 2 * 16 + ln] = (_Float16)(e2 * inv);
    P[row * ldp + 3 * 16 + ln] = (_Float16)(e3 * inv);
  }
}

// ---------------------------------------------------------------------------
// Kernel 0: f32 -> f16 conversion for weights
// ---------------------------------------------------------------------------
__global__ void f2h_kernel(const float* __restrict__ s, _Float16* __restrict__ d,
                           int n) {
  int i = blockIdx.x * 256 + threadIdx.x;
  if (i < n) d[i] = (_Float16)s[i];
}

// ---------------------------------------------------------------------------
// Kernel 1: per-window axis attention (LN + pos -> QKV -> attn -> head QKV)
// One block = one window (64 tokens x 128 ch), 256 threads = 8 waves.
// ---------------------------------------------------------------------------
// LDS arena (halves):
//   qkv_s : [3][4][64][33]     25344 halves   (persists steps 2-3)
//   xs    : [64][136]           8704 halves   (step 1-2, overlaid)
//   pbuf  : [8 waves][16][68]   8704 halves   (step 3, overlaid on xs)
//   xx_s  : [4][64][33]         8448 halves   (step 3-4)
#define LDS_HALVES (25344 + 8704 + 8448)

template <int AXIS>
__global__ __launch_bounds__(256) void axis_attn_kernel(
    const float* __restrict__ x, const float* __restrict__ ln_w,
    const float* __restrict__ ln_b, const float* __restrict__ pos_pre,
    const float* __restrict__ pos_post, const _Float16* __restrict__ wqkv,
    const _Float16* __restrict__ w3, _Float16* __restrict__ oq,
    _Float16* __restrict__ ok, _Float16* __restrict__ ov) {
  __shared__ _Float16 smem[LDS_HALVES];
  _Float16* qkv_s = smem;               // [3][4][64][33]
  _Float16* xs    = smem + 25344;       // [64][136]
  _Float16* pbuf  = smem + 25344;       // [8][16][68]   (reuses xs space)
  _Float16* xx_s  = smem + 25344 + 8704;  // [4][64][33]

  const int w = blockIdx.x;
  const int b = w >> 6;
  const int i = w & 63;
  const int tid = threadIdx.x;
  const int lane = tid & 31, wave = tid >> 5;
  const int hh = lane >> 4, ln = lane & 15;

  __builtin_prefetch(wqkv, 0, 3);

  // ---- Step 1: LayerNorm + positional add, f16 into LDS ------------------
  {
    const int t = tid >> 2;        // token 0..63
    const int part = tid & 3;      // 32-channel slice
    const float* xp =
        x + ((size_t)b * (Ln * Ln) + (AXIS == 0 ? i * Ln + t : t * Ln + i)) * Cn +
        part * 32;
    float vals[32];
    float sum = 0.f, sq = 0.f;
#pragma unroll
    for (int e = 0; e < 32; ++e) {
      float v = xp[e];
      vals[e] = v;
      sum += v;
      sq += v * v;
    }
    sum += __shfl_xor(sum, 1);
    sum += __shfl_xor(sum, 2);
    sq += __shfl_xor(sq, 1);
    sq += __shfl_xor(sq, 2);
    float mean = sum * (1.f / 128.f);
    float var = sq * (1.f / 128.f) - mean * mean;
    float inv = rsqrtf(var + 1e-5f);
    const float* pp = pos_pre + t * Cn + part * 32;
#pragma unroll
    for (int e = 0; e < 32; ++e) {
      int ch = part * 32 + e;
      float v = (vals[e] - mean) * inv * ln_w[ch] + ln_b[ch] + pp[e];
      xs[t * 136 + ch] = (_Float16)v;
    }
  }
  __syncthreads();

  // ---- Step 2: QKV GEMM  [64x128] @ [128x384]  -> qkv_s ------------------
  // 4 m-tiles x 24 n-tiles = 96 tiles; 12 per wave; 4 k-steps of 32.
  for (int tt = 0; tt < 12; ++tt) {
    int tile = wave * 12 + tt;
    int mt = tile / 24, nt = tile % 24;
    v8f acc = zero8();
#pragma unroll
    for (int kk = 0; kk < Cn; kk += 32) {
      v16h a = wmma_load_a(xs, 136, mt * 16, kk, lane);
      v16h bf = load16h(wqkv + (nt * 16 + ln) * Cn + kk + hh * 16);
      acc = wmma_f16(a, bf, acc);
    }
    int c = nt * 16 + ln;            // column in [0,384)
    int which = c >> 7;              // 0=q 1=k 2=v
    int h = (c >> 5) & 3;
    int d = c & 31;
    _Float16* dst = qkv_s + ((which * Hn + h) * Ln) * 33 + d;
#pragma unroll
    for (int r = 0; r < 8; ++r) {
      int tok = mt * 16 + hh * 8 + r;
      dst[tok * 33] = (_Float16)acc[r];
    }
  }
  __syncthreads();

  // ---- Step 3: windowed attention per head, + pos_post -> xx_s -----------
  // 16 tasks (4 heads x 4 row-tiles); 2 per wave. S block stays in VGPRs.
  {
    _Float16* myP = pbuf + wave * 16 * 68;
    for (int it = 0; it < 2; ++it) {
      int task = wave * 2 + it;
      int h = task >> 2, mt = task & 3;
      const _Float16* q_s = qkv_s + (0 * Hn + h) * Ln * 33;
      const _Float16* k_s = qkv_s + (1 * Hn + h) * Ln * 33;
      const _Float16* v_s = qkv_s + (2 * Hn + h) * Ln * 33;

      v16h aq = wmma_load_a(q_s, 33, mt * 16, 0, lane);
      v8f sacc[4];
#pragma unroll
      for (int ntl = 0; ntl < 4; ++ntl) {
        v16h bk = load16h(k_s + (ntl * 16 + ln) * 33 + hh * 16);
        sacc[ntl] = wmma_f16(aq, bk, zero8());
      }
      softmax16x64(sacc, myP, 68, lane);

      v8f oac[2] = {zero8(), zero8()};
#pragma unroll
      for (int kk = 0; kk < Ln; kk += 32) {
        v16h ap = wmma_load_a(myP, 68, 0, kk, lane);
#pragma unroll
        for (int ntl = 0; ntl < 2; ++ntl) {
          v16h bv =
              load16h_strided(v_s + (kk + hh * 16) * 33 + ntl * 16 + ln, 33);
          oac[ntl] = wmma_f16(ap, bv, oac[ntl]);
        }
      }
#pragma unroll
      for (int ntl = 0; ntl < 2; ++ntl) {
        int d = ntl * 16 + ln;
#pragma unroll
        for (int r = 0; r < 8; ++r) {
          int tok = mt * 16 + hh * 8 + r;
          float v = oac[ntl][r] + pos_post[(h * Ln + tok) * HDn + d];
          xx_s[(h * Ln + tok) * 33 + d] = (_Float16)v;
        }
      }
    }
  }
  __syncthreads();

  // ---- Step 4: head QKV GEMM [64x32]@[32x96] per head -> global q/k/v ----
  // 4 heads x 4 m-tiles x 6 n-tiles = 96 tiles; 12 per wave; single k-step.
  for (int tt = 0; tt < 12; ++tt) {
    int tile = wave * 12 + tt;
    int h = tile / 24;
    int rem = tile % 24;
    int mt = rem / 6, nt = rem % 6;
    v16h a = wmma_load_a(xx_s + h * Ln * 33, 33, mt * 16, 0, lane);
    v16h bw = load16h(w3 + (nt * 16 + ln) * HDn + hh * 16);
    v8f acc = wmma_f16(a, bw, zero8());
    int c = nt * 16 + ln;            // [0,96)
    int which = c >> 5;              // 0=q 1=k 2=v
    int d = c & 31;
    _Float16* base = (which == 0) ? oq : (which == 1) ? ok : ov;
    _Float16* dst = base + ((size_t)(w * Hn + h) * Ln) * HDn + d;
#pragma unroll
    for (int r = 0; r < 8; ++r) {
      int tok = mt * 16 + hh * 8 + r;
      dst[tok * HDn] = (_Float16)acc[r];
    }
  }
}

// ---------------------------------------------------------------------------
// Kernel 2: cross attention between axes. One block per window, 8 waves,
// 32 tasks (2 directions x 4 heads x 4 row-tiles), operands straight from
// the f16 global q/k/v streams.
// ---------------------------------------------------------------------------
__global__ __launch_bounds__(256) void cross_kernel(
    const _Float16* __restrict__ xwq, const _Float16* __restrict__ xwk,
    const _Float16* __restrict__ xwv, const _Float16* __restrict__ xhq,
    const _Float16* __restrict__ xhk, const _Float16* __restrict__ xhv,
    _Float16* __restrict__ o1, _Float16* __restrict__ o2) {
  __shared__ _Float16 pbuf[8 * 16 * 68];
  const int w = blockIdx.x;
  const int tid = threadIdx.x;
  const int lane = tid & 31, wave = tid >> 5;
  const int hh = lane >> 4, ln = lane & 15;
  _Float16* myP = pbuf + wave * 16 * 68;

  for (int it = 0; it < 4; ++it) {
    int task = wave * 4 + it;     // [0,32)
    int dir = task >> 4;          // 0: Q=hq,K=wk,V=wv -> o1 ; 1: swap -> o2
    int h = (task >> 2) & 3;
    int mt = task & 3;
    size_t hb = (size_t)(w * Hn + h) * Ln * HDn;
    const _Float16* Q = (dir ? xwq : xhq) + hb;
    const _Float16* K = (dir ? xhk : xwk) + hb;
    const _Float16* V = (dir ? xhv : xwv) + hb;
    _Float16* O = dir ? o2 : o1;

    v16h aq = wmma_load_a(Q, HDn, mt * 16, 0, lane);
    v8f sacc[4];
#pragma unroll
    for (int ntl = 0; ntl < 4; ++ntl) {
      v16h bk = load16h(K + (ntl * 16 + ln) * HDn + hh * 16);
      sacc[ntl] = wmma_f16(aq, bk, zero8());
    }
    softmax16x64(sacc, myP, 68, lane);

    v8f oac[2] = {zero8(), zero8()};
#pragma unroll
    for (int kk = 0; kk < Ln; kk += 32) {
      v16h ap = wmma_load_a(myP, 68, 0, kk, lane);
#pragma unroll
      for (int ntl = 0; ntl < 2; ++ntl) {
        v16h bv =
            load16h_strided(V + (kk + hh * 16) * HDn + ntl * 16 + ln, HDn);
        oac[ntl] = wmma_f16(ap, bv, oac[ntl]);
      }
    }
#pragma unroll
    for (int ntl = 0; ntl < 2; ++ntl) {
      int d = ntl * 16 + ln;
#pragma unroll
      for (int r = 0; r < 8; ++r) {
        int tok = mt * 16 + hh * 8 + r;
        O[((size_t)w * Ln + tok) * Cn + h * HDn + d] = (_Float16)oac[ntl][r];
      }
    }
  }
}

// ---------------------------------------------------------------------------
// Kernel 3: combine (transposed gathers) + residual, f32 output.
// out[b, r*L+c, ch] = o1[(b,c) win, q=r] + o2[(b,r) win, q=c] + x[...]
// ---------------------------------------------------------------------------
__global__ __launch_bounds__(256) void combine_kernel(
    const float* __restrict__ x, const _Float16* __restrict__ o1,
    const _Float16* __restrict__ o2, float* __restrict__ out) {
  size_t idx = (size_t)blockIdx.x * 256 + threadIdx.x;  // B*L*L*C elements
  int ch = idx & 127;
  int c = (idx >> 7) & 63;
  int r = (idx >> 13) & 63;
  int b = (int)(idx >> 19);
  float v = x[idx];
  v += (float)o1[((size_t)(b * Ln + c) * Ln + r) * Cn + ch];
  v += (float)o2[((size_t)(b * Ln + r) * Ln + c) * Cn + ch];
  out[idx] = v;
}

// ---------------------------------------------------------------------------
extern "C" void kernel_launch(void* const* d_in, const int* in_sizes, int n_in,
                              void* d_out, int out_size, void* d_ws,
                              size_t ws_size, hipStream_t stream) {
  (void)in_sizes; (void)n_in; (void)out_size; (void)ws_size;
  const float* x     = (const float*)d_in[0];
  const float* n3_w  = (const float*)d_in[1];
  const float* n3_b  = (const float*)d_in[2];
  const float* n4_w  = (const float*)d_in[3];
  const float* n4_b  = (const float*)d_in[4];
  const float* wqkv1 = (const float*)d_in[5];
  const float* wqkv2 = (const float*)d_in[6];
  const float* wqkv3 = (const float*)d_in[7];
  const float* wqkv4 = (const float*)d_in[8];
  const float* pos1  = (const float*)d_in[9];
  const float* pos2  = (const float*)d_in[10];
  const float* pos3  = (const float*)d_in[11];
  const float* pos4  = (const float*)d_in[12];
  float* out = (float*)d_out;

  // Workspace carving (256B aligned).
  char* p = (char*)d_ws;
  auto alloc = [&](size_t bytes) {
    void* r = (void*)p;
    p += (bytes + 255) & ~(size_t)255;
    return r;
  };
  const size_t WQ_BIG = 3 * Cn * Cn;      // 49152 halves
  const size_t WQ_SM  = 3 * HDn * HDn;    // 3072 halves
  const size_t QKVBUF = (size_t)NWIN * Hn * Ln * HDn;   // 16.7M halves
  const size_t OBUF   = (size_t)NWIN * Ln * Cn;         // 16.7M halves

  _Float16* wq1h = (_Float16*)alloc(WQ_BIG * 2);
  _Float16* wq2h = (_Float16*)alloc(WQ_BIG * 2);
  _Float16* wq3h = (_Float16*)alloc(WQ_SM * 2);
  _Float16* wq4h = (_Float16*)alloc(WQ_SM * 2);
  _Float16* xwq  = (_Float16*)alloc(QKVBUF * 2);
  _Float16* xwk  = (_Float16*)alloc(QKVBUF * 2);
  _Float16* xwv  = (_Float16*)alloc(QKVBUF * 2);
  _Float16* xhq  = (_Float16*)alloc(QKVBUF * 2);
  _Float16* xhk  = (_Float16*)alloc(QKVBUF * 2);
  _Float16* xhv  = (_Float16*)alloc(QKVBUF * 2);
  _Float16* o1b  = (_Float16*)alloc(OBUF * 2);
  _Float16* o2b  = (_Float16*)alloc(OBUF * 2);

  // 0) weight conversion
  f2h_kernel<<<(WQ_BIG + 255) / 256, 256, 0, stream>>>(wqkv1, wq1h, (int)WQ_BIG);
  f2h_kernel<<<(WQ_BIG + 255) / 256, 256, 0, stream>>>(wqkv2, wq2h, (int)WQ_BIG);
  f2h_kernel<<<(WQ_SM + 255) / 256, 256, 0, stream>>>(wqkv3, wq3h, (int)WQ_SM);
  f2h_kernel<<<(WQ_SM + 255) / 256, 256, 0, stream>>>(wqkv4, wq4h, (int)WQ_SM);

  // 1) row-axis and 2) column-axis streams
  axis_attn_kernel<0><<<NWIN, 256, 0, stream>>>(x, n3_w, n3_b, pos1, pos3,
                                                wq1h, wq3h, xwq, xwk, xwv);
  axis_attn_kernel<1><<<NWIN, 256, 0, stream>>>(x, n4_w, n4_b, pos2, pos4,
                                                wq2h, wq4h, xhq, xhk, xhv);

  // 3) cross attention
  cross_kernel<<<NWIN, 256, 0, stream>>>(xwq, xwk, xwv, xhq, xhk, xhv, o1b,
                                         o2b);

  // 4) combine + residual
  const size_t total = (size_t)Bn * Ln * Ln * Cn;  // 16,777,216
  combine_kernel<<<(unsigned)(total / 256), 256, 0, stream>>>(x, o1b, o2b, out);
}
